// RelationalMessagePassingNeuralNetwork_64424509440348
// MI455X (gfx1250) — matrix operand order repaired and checked by reading, not compiled, not run on hardware
//
#include <hip/hip_runtime.h>

// ---------------------------------------------------------------------------
// Relational message-passing GNN for MI455X (gfx1250, wave32, WMMA).
//   edge:  y_e = W2_t( relu( W1_t [ef | nf[src] | nf[dst]] + b1 ) ) + b2
//   agg:   agg[dst[j]] += new_ef[j]           (fused atomics, L2 resident)
//   node:  y_n = W2_t( relu( W1_t [agg | nf] + b1 ) ) + b2
// Reference quirk out[perm[perm]] handled via stable type sort (perm/pos).
// GEMMs: bf16 inputs, f32 accumulate, v_wmma_f32_16x16x32_bf16.
// Weight panels double-buffered through LDS (regs->LDS ping-pong, 1 barrier
// per K-panel); B fragments preloaded so WMMAs issue back-to-back.
// All f32->bf16 conversion via vector convert -> packed v_cvt_pk_bf16_f32.
// ---------------------------------------------------------------------------

#define DFEAT 128
#define WST 40    // LDS B-panel stride (u16), padded, 80B (16B aligned)
#define HST 136   // LDS hidden stride (u16), padded, 272B (16B aligned)
#define IPT 32    // elements per thread-slot in the stable counting sort

typedef __attribute__((ext_vector_type(16))) __bf16 v16bf;
typedef __attribute__((ext_vector_type(2)))  __bf16 v2bf;
typedef __attribute__((ext_vector_type(16))) float  v16f;
typedef __attribute__((ext_vector_type(2)))  float  v2f;
typedef __attribute__((ext_vector_type(8)))  float  v8f;
typedef __attribute__((ext_vector_type(8)))  unsigned short v8us;

union BFV { v16bf v; unsigned short s[16]; v8us h[2]; };

// Single f32->bf16 (RNE) for scalar stores.
__device__ __forceinline__ unsigned short f2bf(float f) {
  union { __bf16 b; unsigned short u; } c;
  c.b = (__bf16)f;
  return c.u;
}
// Paired f32->bf16: one v_cvt_pk_bf16_f32, no repacking.
__device__ __forceinline__ unsigned pack2(float lo, float hi) {
  v2f f; f[0] = lo; f[1] = hi;
  union { v2bf b; unsigned u; } c;
  c.b = __builtin_convertvector(f, v2bf);
  return c.u;
}

// A fragment (16x32 bf16): lanes 0-15 hold row M=lane, K {0..7,16..23};
// lanes 16-31 hold row M=lane-16, K {8..15,24..31}.  koff = lane<16 ? 0 : 8.
// Vector convert: 8x v_cvt_pk_bf16_f32 straight into the fragment registers.
__device__ __forceinline__ v16bf ld_a_global(const float* rowp, int koff) {
  const float4* p0 = (const float4*)(rowp + koff);
  const float4* p1 = (const float4*)(rowp + koff + 16);
  float4 a0 = p0[0], a1 = p0[1], a2 = p1[0], a3 = p1[1];
  v16f f;
  f[0]  = a0.x; f[1]  = a0.y; f[2]  = a0.z; f[3]  = a0.w;
  f[4]  = a1.x; f[5]  = a1.y; f[6]  = a1.z; f[7]  = a1.w;
  f[8]  = a2.x; f[9]  = a2.y; f[10] = a2.z; f[11] = a2.w;
  f[12] = a3.x; f[13] = a3.y; f[14] = a3.z; f[15] = a3.w;
  return __builtin_convertvector(f, v16bf);
}

// Weight-panel staging, split into load (global->regs) and store (regs->LDS)
// halves so the global loads of panel k+1 overlap the WMMAs of panel k.
struct StageRegs { float4 w0[4]; float4 w1[4]; };

__device__ __forceinline__ void stage_load(const float* __restrict__ Wsrc,
                                           int tid, StageRegs& r) {
  const float4* wp = (const float4*)Wsrc;
#pragma unroll
  for (int it = 0; it < 4; ++it) {
    int t2 = it * 128 + tid;               // 0..511
    int k2 = t2 >> 5;                      // k = 2*k2
    int nq = t2 & 31;                      // n quad
    r.w0[it] = wp[(2 * k2) * 32 + nq];
    r.w1[it] = wp[(2 * k2 + 1) * 32 + nq];
  }
}
__device__ __forceinline__ void stage_store(unsigned short* Wbuf, int tid,
                                            const StageRegs& r) {
#pragma unroll
  for (int it = 0; it < 4; ++it) {
    int t2 = it * 128 + tid;
    int k2 = t2 >> 5;
    int nq = t2 & 31;
    int n4 = nq * 4, k = 2 * k2;
    *(unsigned*)&Wbuf[(size_t)(n4 + 0) * WST + k] = pack2(r.w0[it].x, r.w1[it].x);
    *(unsigned*)&Wbuf[(size_t)(n4 + 1) * WST + k] = pack2(r.w0[it].y, r.w1[it].y);
    *(unsigned*)&Wbuf[(size_t)(n4 + 2) * WST + k] = pack2(r.w0[it].z, r.w1[it].z);
    *(unsigned*)&Wbuf[(size_t)(n4 + 3) * WST + k] = pack2(r.w0[it].w, r.w1[it].w);
  }
}

// B fragment (32x16 bf16): lanes 0-15 hold col N=lane, K {0..7,16..23};
// lanes 16-31 hold col N=lane-16, K {8..15,24..31} (matches A fragment).
__device__ __forceinline__ v16bf ld_b_lds(const unsigned short* Wbuf, int col, int lane) {
  int kb = (lane < 16) ? 0 : 8;
  BFV b;
  b.h[0] = *(const v8us*)&Wbuf[(size_t)col * WST + kb];
  b.h[1] = *(const v8us*)&Wbuf[(size_t)col * WST + kb + 16];
  return b.v;
}

// --------------------------- helper kernels --------------------------------

__global__ void zero_f32_kernel(float* p, long long n) {
  long long i = (long long)blockIdx.x * blockDim.x + threadIdx.x;
  long long stride = (long long)gridDim.x * blockDim.x;
  for (; i < n; i += stride) p[i] = 0.0f;
}

// Stable counting sort by type, 3 deterministic passes. tcnt is [slots][4].
__global__ void type_count_kernel(const int* __restrict__ tvec, int n,
                                  int* __restrict__ tcnt, int slots) {
  int s = blockIdx.x * blockDim.x + threadIdx.x;
  if (s >= slots) return;
  int c[4] = {0, 0, 0, 0};
  int b = s * IPT;
  int e = b + IPT; if (e > n) e = n;
  for (int i = b; i < e; ++i) c[tvec[i]]++;
#pragma unroll
  for (int t = 0; t < 4; ++t) tcnt[s * 4 + t] = c[t];
}

__global__ void type_scan_kernel(int* __restrict__ tcnt, int slots, int T) {
  __shared__ long long tot[4];
  int t = threadIdx.x;
  if (t < T) {
    long long sum = 0;
    for (int s = 0; s < slots; ++s) sum += tcnt[s * 4 + t];
    tot[t] = sum;
  }
  __syncthreads();
  if (t == 0) {
    long long off = 0;
    for (int k = 0; k < T; ++k) { long long c = tot[k]; tot[k] = off; off += c; }
  }
  __syncthreads();
  if (t < T) {
    long long run = tot[t];
    for (int s = 0; s < slots; ++s) {
      int c = tcnt[s * 4 + t];
      tcnt[s * 4 + t] = (int)run;
      run += c;
    }
  }
}

__global__ void type_rank_kernel(const int* __restrict__ tvec, int n,
                                 const int* __restrict__ tcnt, int slots,
                                 int* __restrict__ perm, int* __restrict__ pos) {
  int s = blockIdx.x * blockDim.x + threadIdx.x;
  if (s >= slots) return;
  int base[4];
#pragma unroll
  for (int t = 0; t < 4; ++t) base[t] = tcnt[s * 4 + t];
  int b = s * IPT;
  int e = b + IPT; if (e > n) e = n;
  for (int i = b; i < e; ++i) {
    int ty = tvec[i];
    int p = base[ty]++;
    pos[i] = p;       // position of element i in the stable sort
    perm[p] = i;      // element at position p
  }
}

// ----------------------------- edge MLP ------------------------------------
// Block = 128 threads = 4 wave32s; 64 rows x 128 cols per block.
// Rows are sorted positions p: element i = perm[p]; output row = pos[p].

__global__ __launch_bounds__(128, 1)
void edge_mlp_kernel(const float* __restrict__ nf, const float* __restrict__ ef,
                     const int* __restrict__ src, const int* __restrict__ dst,
                     const int* __restrict__ etype,
                     const float* __restrict__ Wh, const float* __restrict__ bh,
                     const float* __restrict__ Wo, const float* __restrict__ bo,
                     const int* __restrict__ perm, const int* __restrict__ pos,
                     float* __restrict__ outEdge, float* __restrict__ agg,
                     int E, int T) {
  __shared__ __align__(16) unsigned short Wbuf[2][128 * WST];
  __shared__ __align__(16) unsigned short Hbuf[64 * HST];
  __shared__ int eS[64], sS[64], dS[64], aS[64], oS[64], tyS[64];
  __shared__ unsigned tmask;

  const int tid = threadIdx.x;
  const int wid = tid >> 5;
  const int lane = tid & 31;
  const int p0 = blockIdx.x * 64;

  if (tid == 0) tmask = 0u;
  __syncthreads();
  if (tid < 64) {
    int p = p0 + tid;
    int ty = -1, e = 0, sr = 0, dr = 0, orow = 0, ad = 0;
    if (p < E) {
      e = perm[p]; ty = etype[e];
      sr = src[e]; dr = dst[e];
      orow = pos[p];          // output row (perm[perm] gather quirk)
      ad = dst[orow];         // aggregation target follows output row
      atomicOr(&tmask, 1u << ty);
    }
    eS[tid] = e; sS[tid] = sr; dS[tid] = dr;
    oS[tid] = orow; aS[tid] = ad; tyS[tid] = ty;
  }
  __syncthreads();

  const int lr = wid * 16 + (lane & 15);
  const float* rowEf  = ef + (size_t)eS[lr] * DFEAT;
  const float* rowSrc = nf + (size_t)sS[lr] * DFEAT;
  const float* rowDst = nf + (size_t)dS[lr] * DFEAT;
  const int koff = (lane < 16) ? 0 : 8;
  const int mycol = lane & 15;
  const int rbase = wid * 16 + ((lane >> 4) << 3);

  for (int t = 0; t < T; ++t) {
    if (!((tmask >> t) & 1u)) continue;
    const float* WhT = Wh + (size_t)t * 384 * DFEAT;
    const float* WoT = Wo + (size_t)t * 128 * DFEAT;

    v8f acc[8];
#pragma unroll
    for (int n = 0; n < 8; ++n)
#pragma unroll
      for (int r = 0; r < 8; ++r) acc[n][r] = 0.0f;

    // ---- layer 1: [64x384] x [384x128], 12 K-panels of 32, ping-pong ----
    StageRegs sr;
    stage_load(WhT, tid, sr);
    stage_store(Wbuf[0], tid, sr);
    int cur = 0;
    for (int kp = 0; kp < 12; ++kp) {
      __syncthreads();   // panel `cur` visible; other buffer free to overwrite
      if (kp + 2 < 12) __builtin_prefetch(WhT + (size_t)(kp + 2) * 32 * DFEAT, 0, 1);
      if (kp + 1 < 12) stage_load(WhT + (size_t)(kp + 1) * 32 * DFEAT, tid, sr);
      const float* rp = (kp < 4) ? rowEf : (kp < 8 ? rowSrc : rowDst);
      v16bf a = ld_a_global(rp + (kp & 3) * 32, koff);
      const unsigned short* wb = Wbuf[cur];
      v16bf bf[8];
#pragma unroll
      for (int n = 0; n < 8; ++n) bf[n] = ld_b_lds(wb, n * 16 + mycol, lane);
#pragma unroll
      for (int n = 0; n < 8; ++n)
        acc[n] = __builtin_amdgcn_wmma_f32_16x16x32_bf16(
            false, a, false, bf[n], (short)0, acc[n], false, false);
      if (kp + 1 < 12) stage_store(Wbuf[cur ^ 1], tid, sr);
      cur ^= 1;
    }

    // bias + relu -> bf16 hidden in LDS
#pragma unroll
    for (int n = 0; n < 8; ++n) {
      int col = n * 16 + mycol;
      float bias = bh[t * 128 + col];
#pragma unroll
      for (int r = 0; r < 8; ++r) {
        float v = acc[n][r] + bias;
        v = v > 0.0f ? v : 0.0f;
        Hbuf[(size_t)(rbase + r) * HST + col] = f2bf(v);
      }
    }
    __syncthreads();

    // ---- layer 2: [64x128] x [128x128], 4 K-panels, ping-pong ----
    v8f acc2[8];
#pragma unroll
    for (int n = 0; n < 8; ++n)
#pragma unroll
      for (int r = 0; r < 8; ++r) acc2[n][r] = 0.0f;

    stage_load(WoT, tid, sr);
    stage_store(Wbuf[0], tid, sr);
    cur = 0;
    for (int kp = 0; kp < 4; ++kp) {
      __syncthreads();
      if (kp + 1 < 4) stage_load(WoT + (size_t)(kp + 1) * 32 * DFEAT, tid, sr);
      int kb = kp * 32;
      BFV aa;
      aa.h[0] = *(const v8us*)&Hbuf[(size_t)(wid * 16 + mycol) * HST + kb + koff];
      aa.h[1] = *(const v8us*)&Hbuf[(size_t)(wid * 16 + mycol) * HST + kb + koff + 16];
      const unsigned short* wb = Wbuf[cur];
      v16bf bf[8];
#pragma unroll
      for (int n = 0; n < 8; ++n) bf[n] = ld_b_lds(wb, n * 16 + mycol, lane);
#pragma unroll
      for (int n = 0; n < 8; ++n)
        acc2[n] = __builtin_amdgcn_wmma_f32_16x16x32_bf16(
            false, aa.v, false, bf[n], (short)0, acc2[n], false, false);
      if (kp + 1 < 4) stage_store(Wbuf[cur ^ 1], tid, sr);
      cur ^= 1;
    }

    // epilogue: write new_ef + fused segment-sum atomics
#pragma unroll
    for (int n = 0; n < 8; ++n) {
      int col = n * 16 + mycol;
      float bias = bo[t * 128 + col];
#pragma unroll
      for (int r = 0; r < 8; ++r) {
        int row = rbase + r;
        if (tyS[row] == t) {
          float v = acc2[n][r] + bias;
          outEdge[(size_t)oS[row] * DFEAT + col] = v;
          atomicAdd(&agg[(size_t)aS[row] * DFEAT + col], v);
        }
      }
    }
    __syncthreads();   // next t iteration re-stages Wbuf / rewrites Hbuf
  }
}

// ----------------------------- node MLP ------------------------------------

__global__ __launch_bounds__(128, 1)
void node_mlp_kernel(const float* __restrict__ nf, const float* __restrict__ agg,
                     const int* __restrict__ ntype,
                     const float* __restrict__ Wh, const float* __restrict__ bh,
                     const float* __restrict__ Wo, const float* __restrict__ bo,
                     const int* __restrict__ perm, const int* __restrict__ pos,
                     float* __restrict__ outNode, int N, int T) {
  __shared__ __align__(16) unsigned short Wbuf[2][128 * WST];
  __shared__ __align__(16) unsigned short Hbuf[64 * HST];
  __shared__ int iS[64], oS[64], tyS[64];
  __shared__ unsigned tmask;

  const int tid = threadIdx.x;
  const int wid = tid >> 5;
  const int lane = tid & 31;
  const int p0 = blockIdx.x * 64;

  if (tid == 0) tmask = 0u;
  __syncthreads();
  if (tid < 64) {
    int p = p0 + tid;
    int ty = -1, i = 0, orow = 0;
    if (p < N) {
      i = perm[p]; ty = ntype[i];
      orow = pos[p];
      atomicOr(&tmask, 1u << ty);
    }
    iS[tid] = i; oS[tid] = orow; tyS[tid] = ty;
  }
  __syncthreads();

  const int lr = wid * 16 + (lane & 15);
  const float* rowAgg = agg + (size_t)iS[lr] * DFEAT;
  const float* rowNf  = nf + (size_t)iS[lr] * DFEAT;
  const int koff = (lane < 16) ? 0 : 8;
  const int mycol = lane & 15;
  const int rbase = wid * 16 + ((lane >> 4) << 3);

  for (int t = 0; t < T; ++t) {
    if (!((tmask >> t) & 1u)) continue;
    const float* WhT = Wh + (size_t)t * 256 * DFEAT;
    const float* WoT = Wo + (size_t)t * 128 * DFEAT;

    v8f acc[8];
#pragma unroll
    for (int n = 0; n < 8; ++n)
#pragma unroll
      for (int r = 0; r < 8; ++r) acc[n][r] = 0.0f;

    // ---- layer 1: [64x256] x [256x128], 8 K-panels, ping-pong ----
    StageRegs sr;
    stage_load(WhT, tid, sr);
    stage_store(Wbuf[0], tid, sr);
    int cur = 0;
    for (int kp = 0; kp < 8; ++kp) {
      __syncthreads();
      if (kp + 2 < 8) __builtin_prefetch(WhT + (size_t)(kp + 2) * 32 * DFEAT, 0, 1);
      if (kp + 1 < 8) stage_load(WhT + (size_t)(kp + 1) * 32 * DFEAT, tid, sr);
      const float* rp = (kp < 4) ? rowAgg : rowNf;
      v16bf a = ld_a_global(rp + (kp & 3) * 32, koff);
      const unsigned short* wb = Wbuf[cur];
      v16bf bf[8];
#pragma unroll
      for (int n = 0; n < 8; ++n) bf[n] = ld_b_lds(wb, n * 16 + mycol, lane);
#pragma unroll
      for (int n = 0; n < 8; ++n)
        acc[n] = __builtin_amdgcn_wmma_f32_16x16x32_bf16(
            false, a, false, bf[n], (short)0, acc[n], false, false);
      if (kp + 1 < 8) stage_store(Wbuf[cur ^ 1], tid, sr);
      cur ^= 1;
    }

#pragma unroll
    for (int n = 0; n < 8; ++n) {
      int col = n * 16 + mycol;
      float bias = bh[t * 128 + col];
#pragma unroll
      for (int r = 0; r < 8; ++r) {
        float v = acc[n][r] + bias;
        v = v > 0.0f ? v : 0.0f;
        Hbuf[(size_t)(rbase + r) * HST + col] = f2bf(v);
      }
    }
    __syncthreads();

    // ---- layer 2: [64x128] x [128x128], 4 K-panels, ping-pong ----
    v8f acc2[8];
#pragma unroll
    for (int n = 0; n < 8; ++n)
#pragma unroll
      for (int r = 0; r < 8; ++r) acc2[n][r] = 0.0f;

    stage_load(WoT, tid, sr);
    stage_store(Wbuf[0], tid, sr);
    cur = 0;
    for (int kp = 0; kp < 4; ++kp) {
      __syncthreads();
      if (kp + 1 < 4) stage_load(WoT + (size_t)(kp + 1) * 32 * DFEAT, tid, sr);
      int kb = kp * 32;
      BFV aa;
      aa.h[0] = *(const v8us*)&Hbuf[(size_t)(wid * 16 + mycol) * HST + kb + koff];
      aa.h[1] = *(const v8us*)&Hbuf[(size_t)(wid * 16 + mycol) * HST + kb + koff + 16];
      const unsigned short* wb = Wbuf[cur];
      v16bf bf[8];
#pragma unroll
      for (int n = 0; n < 8; ++n) bf[n] = ld_b_lds(wb, n * 16 + mycol, lane);
#pragma unroll
      for (int n = 0; n < 8; ++n)
        acc2[n] = __builtin_amdgcn_wmma_f32_16x16x32_bf16(
            false, aa.v, false, bf[n], (short)0, acc2[n], false, false);
      if (kp + 1 < 4) stage_store(Wbuf[cur ^ 1], tid, sr);
      cur ^= 1;
    }

#pragma unroll
    for (int n = 0; n < 8; ++n) {
      int col = n * 16 + mycol;
      float bias = bo[t * 128 + col];
#pragma unroll
      for (int r = 0; r < 8; ++r) {
        int row = rbase + r;
        if (tyS[row] == t) {
          outNode[(size_t)oS[row] * DFEAT + col] = acc2[n][r] + bias;
        }
      }
    }
    __syncthreads();
  }
}

// ------------------------------ launch -------------------------------------

extern "C" void kernel_launch(void* const* d_in, const int* in_sizes, int n_in,
                              void* d_out, int out_size, void* d_ws, size_t ws_size,
                              hipStream_t stream) {
  const float* nf    = (const float*)d_in[0];
  const float* ef    = (const float*)d_in[1];
  const int*   src   = (const int*)d_in[2];
  const int*   dst   = (const int*)d_in[3];
  const int*   etype = (const int*)d_in[4];
  const int*   ntype = (const int*)d_in[5];
  const float* Weh   = (const float*)d_in[6];
  const float* beh   = (const float*)d_in[7];
  const float* Weo   = (const float*)d_in[8];
  const float* beo   = (const float*)d_in[9];
  const float* Wnh   = (const float*)d_in[10];
  const float* bnh   = (const float*)d_in[11];
  const float* Wno   = (const float*)d_in[12];
  const float* bno   = (const float*)d_in[13];

  const int N  = in_sizes[0] / DFEAT;
  const int E  = in_sizes[1] / DFEAT;
  const int TE = in_sizes[6] / (3 * DFEAT * DFEAT);
  const int TN = in_sizes[10] / (2 * DFEAT * DFEAT);

  float* outNode = (float*)d_out;
  float* outEdge = outNode + (size_t)N * DFEAT;

  // workspace carve-out (256B aligned)
  char* ws = (char*)d_ws;
  auto carve = [&](size_t bytes) {
    char* p = ws;
    ws += (bytes + 255) & ~(size_t)255;
    return p;
  };
  float* agg    = (float*)carve((size_t)N * DFEAT * sizeof(float));
  int* perm_e   = (int*)carve((size_t)E * sizeof(int));
  int* pos_e    = (int*)carve((size_t)E * sizeof(int));
  int* perm_n   = (int*)carve((size_t)N * sizeof(int));
  int* pos_n    = (int*)carve((size_t)N * sizeof(int));
  const int slots_e = (E + IPT - 1) / IPT;
  const int slots_n = (N + IPT - 1) / IPT;
  int* tcnt_e   = (int*)carve((size_t)slots_e * 4 * sizeof(int));
  int* tcnt_n   = (int*)carve((size_t)slots_n * 4 * sizeof(int));

  // 1) zero aggregation buffer (accumulated every call)
  zero_f32_kernel<<<2048, 256, 0, stream>>>(agg, (long long)N * DFEAT);

  // 2) stable type sorts (deterministic, no atomics in ranking path)
  type_count_kernel<<<(slots_e + 255) / 256, 256, 0, stream>>>(etype, E, tcnt_e, slots_e);
  type_scan_kernel<<<1, 64, 0, stream>>>(tcnt_e, slots_e, TE);
  type_rank_kernel<<<(slots_e + 255) / 256, 256, 0, stream>>>(etype, E, tcnt_e, slots_e, perm_e, pos_e);

  type_count_kernel<<<(slots_n + 255) / 256, 256, 0, stream>>>(ntype, N, tcnt_n, slots_n);
  type_scan_kernel<<<1, 64, 0, stream>>>(tcnt_n, slots_n, TN);
  type_rank_kernel<<<(slots_n + 255) / 256, 256, 0, stream>>>(ntype, N, tcnt_n, slots_n, perm_n, pos_n);

  // 3) edge MLP (WMMA) + fused segment-sum into agg
  edge_mlp_kernel<<<(E + 63) / 64, 128, 0, stream>>>(
      nf, ef, src, dst, etype, Weh, beh, Weo, beo, perm_e, pos_e,
      outEdge, agg, E, TE);

  // 4) node MLP (WMMA)
  node_mlp_kernel<<<(N + 63) / 64, 128, 0, stream>>>(
      nf, agg, ntype, Wnh, bnh, Wno, bno, perm_n, pos_n,
      outNode, N, TN);

  (void)n_in; (void)out_size; (void)ws_size;
}
